// Graph_branch_26920855011374
// MI455X (gfx1250) — compile-verified, matrix operand order checked by reading, not run on hardware
//
#include <hip/hip_runtime.h>
#include <hip/hip_bf16.h>

// ---------------------------------------------------------------------------
// GAT-style block attention + MLP + LayerNorm, CDNA5 (gfx1250) WMMA version.
// B=4, N=4096, FIN=FOUT=256, k=64, C=64.
// ---------------------------------------------------------------------------

typedef _Float16 half8_t  __attribute__((ext_vector_type(8)));
typedef _Float16 half16_t __attribute__((ext_vector_type(16)));
typedef float    float8_t __attribute__((ext_vector_type(8)));

#define ALPHA_ATT 0.2f
#define ALPHA_MLP 0.01f
#define LN_EPS    1e-5f

// ---- fragment loaders -----------------------------------------------------
// A-fragment (16x32 f16, M x K): lane<16 holds row M=lane%16, half K pattern
// {k0..k0+7, k0+16..k0+23}; lane>=16 same row set with +8 K offset.
// B-fragment is symmetric (lane -> N column, "loaded column-major"), i.e. we
// read it from a transposed (N-major) buffer with the identical pattern.
// q must already point at  base + row*ld + k0 + ((lane<16)?0:8).

__device__ __forceinline__ half16_t load_frag_h(const _Float16* q) {
  half16_t f;
  ((half8_t*)&f)[0] = *(const half8_t*)(q);
  ((half8_t*)&f)[1] = *(const half8_t*)(q + 16);
  return f;
}

__device__ __forceinline__ half16_t load_frag_f32(const float* q) {
  float4 x0 = *(const float4*)(q);
  float4 x1 = *(const float4*)(q + 4);
  float4 x2 = *(const float4*)(q + 16);
  float4 x3 = *(const float4*)(q + 20);
  half16_t f;
  f[0]=(_Float16)x0.x;  f[1]=(_Float16)x0.y;  f[2]=(_Float16)x0.z;  f[3]=(_Float16)x0.w;
  f[4]=(_Float16)x1.x;  f[5]=(_Float16)x1.y;  f[6]=(_Float16)x1.z;  f[7]=(_Float16)x1.w;
  f[8]=(_Float16)x2.x;  f[9]=(_Float16)x2.y;  f[10]=(_Float16)x2.z; f[11]=(_Float16)x2.w;
  f[12]=(_Float16)x3.x; f[13]=(_Float16)x3.y; f[14]=(_Float16)x3.z; f[15]=(_Float16)x3.w;
  return f;
}

__device__ __forceinline__ float8_t wmma_f16(half16_t a, half16_t b, float8_t c) {
  return __builtin_amdgcn_wmma_f32_16x16x32_f16(false, a, false, b, (short)0, c,
                                                false, false);
}

// ---------------------------------------------------------------------------
// K0: convert + transpose the four 256x256 weight matrices to f16 (N-major).
// ---------------------------------------------------------------------------
__global__ __launch_bounds__(256) void prep_weights_k(
    const float* __restrict__ W,  const float* __restrict__ W1,
    const float* __restrict__ W2, const float* __restrict__ W3,
    _Float16* __restrict__ WT,  _Float16* __restrict__ W1T,
    _Float16* __restrict__ W2T, _Float16* __restrict__ W3T) {
  int idx = blockIdx.x * 256 + threadIdx.x;   // 0 .. 4*65536-1
  int m = idx >> 16;
  int r = idx & 65535;
  int n = r >> 8, k = r & 255;
  const float* src = (m == 0) ? W : (m == 1) ? W1 : (m == 2) ? W2 : W3;
  _Float16*    dst = (m == 0) ? WT : (m == 1) ? W1T : (m == 2) ? W2T : W3T;
  dst[n * 256 + k] = (_Float16)src[k * 256 + n];
}

// ---------------------------------------------------------------------------
// K1: Wh = h @ W   (M=16384, N=256, K=256), one 16x16 tile per wave.
// ---------------------------------------------------------------------------
__global__ __launch_bounds__(256) void gemm_wh_k(
    const float* __restrict__ h, const _Float16* __restrict__ WT,
    float* __restrict__ Wh) {
  const int lane = threadIdx.x & 31;
  const int wave = threadIdx.x >> 5;
  const int tile = blockIdx.x * 8 + wave;     // 16384 tiles
  const int tm = tile >> 4;                   // 0..1023
  const int tn = tile & 15;                   // 0..15
  const int koff = (lane < 16) ? 0 : 8;

  const float*    ap = h  + (size_t)(tm * 16 + (lane & 15)) * 256 + koff;
  const _Float16* bp = WT + (size_t)(tn * 16 + (lane & 15)) * 256 + koff;

  float8_t acc = {};
#pragma unroll
  for (int kk = 0; kk < 8; ++kk) {
    half16_t a = load_frag_f32(ap + kk * 32);
    half16_t b = load_frag_h(bp + kk * 32);
    acc = wmma_f16(a, b, acc);
  }

  const int n     = tn * 16 + (lane & 15);
  const int mbase = tm * 16 + ((lane < 16) ? 0 : 8);
#pragma unroll
  for (int r = 0; r < 8; ++r)
    Wh[(size_t)(mbase + r) * 256 + n] = acc[r];
}

// ---------------------------------------------------------------------------
// K2: per (b,c) 64-block: e = leaky(Wh.a, 0.2); softmax(adj_blk * e[i]) rows;
//     out = scores @ Whc  (WMMA from LDS).
// ---------------------------------------------------------------------------
#define WHT_LD 72   // 64 + 8 halves padding (16B aligned rows)
#define SCO_LD 72

__global__ __launch_bounds__(256) void attn_k(
    const float* __restrict__ Wh, const float* __restrict__ adj,
    const float* __restrict__ av, float* __restrict__ attn) {
  __shared__ _Float16 whT[256 * WHT_LD];   // Whc^T : [f][i]
  __shared__ _Float16 scoA[64 * SCO_LD];   // scores: [i][j]
  __shared__ float    e_s[64];

  const int tid = threadIdx.x;
  const int b = blockIdx.x >> 6;
  const int c = blockIdx.x & 63;
  const int rowbase = b * 4096 + c * 64;

  // stage Whc^T into LDS (f16); iteration 'it' reads one full row (coalesced)
  for (int it = 0; it < 64; ++it) {
    float v = Wh[(size_t)(rowbase + it) * 256 + tid];
    whT[tid * WHT_LD + it] = (_Float16)v;
  }

  // e[i] = leaky(Wh_row_i . a, 0.2)  in f32
  if (tid < 64) {
    const float* row = Wh + (size_t)(rowbase + tid) * 256;
    float s = 0.f;
    for (int f = 0; f < 256; ++f) s += row[f] * av[f];
    e_s[tid] = (s >= 0.f) ? s : ALPHA_ATT * s;
  }
  __syncthreads();

  // softmax over j of adj_blk[i][j] * e[i]; 4 lanes per row, 16 cols each
  {
    const int i = tid >> 2;
    const int part = tid & 3;
    const float ei = e_s[i];
    const size_t arow = (size_t)b * 4096 * 4096 + (size_t)(c * 64 + i) * 4096 +
                        (size_t)(c * 64) + part * 16;
    float sv[16];
    float mx = -3.4e38f;
#pragma unroll
    for (int q = 0; q < 16; ++q) {
      float v = adj[arow + q] * ei;
      sv[q] = v;
      mx = fmaxf(mx, v);
    }
    mx = fmaxf(mx, __shfl_xor(mx, 1, 4));
    mx = fmaxf(mx, __shfl_xor(mx, 2, 4));
    float ssum = 0.f;
#pragma unroll
    for (int q = 0; q < 16; ++q) { sv[q] = __expf(sv[q] - mx); ssum += sv[q]; }
    ssum += __shfl_xor(ssum, 1, 4);
    ssum += __shfl_xor(ssum, 2, 4);
    const float inv = 1.f / ssum;
#pragma unroll
    for (int q = 0; q < 16; ++q)
      scoA[i * SCO_LD + part * 16 + q] = (_Float16)(sv[q] * inv);
  }
  __syncthreads();

  // out = scores(64x64) @ Whc(64x256): 64 tiles, 8 per wave, K-loop of 2
  const int lane = tid & 31;
  const int wave = tid >> 5;
  const int koff = (lane < 16) ? 0 : 8;
  for (int tt = 0; tt < 8; ++tt) {
    const int tile = wave * 8 + tt;
    const int tm = tile >> 4, tn = tile & 15;
    const _Float16* ap = scoA + (tm * 16 + (lane & 15)) * SCO_LD + koff;
    const _Float16* bp = whT  + (tn * 16 + (lane & 15)) * WHT_LD + koff;
    float8_t acc = {};
    acc = wmma_f16(load_frag_h(ap),      load_frag_h(bp),      acc);
    acc = wmma_f16(load_frag_h(ap + 32), load_frag_h(bp + 32), acc);
    const int n = tn * 16 + (lane & 15);
    const int mrow = rowbase + tm * 16 + ((lane < 16) ? 0 : 8);
#pragma unroll
    for (int r = 0; r < 8; ++r)
      attn[(size_t)(mrow + r) * 256 + n] = acc[r];
  }
}

// ---------------------------------------------------------------------------
// K3: 3-layer MLP (LeakyReLU 0.01 after first two) + residual + LayerNorm.
// One wave owns 16 rows; A fragments round-trip through a per-wave LDS stage.
// ---------------------------------------------------------------------------
#define STAGE_LD 272   // 256 + 16 halves pad (16B aligned rows)

__device__ __forceinline__ void mlp_layer(const _Float16* stageA,
                                          const _Float16* Wt, int lane,
                                          float8_t accs[16]) {
  const int koff = (lane < 16) ? 0 : 8;
  const _Float16* ap0 = stageA + (lane & 15) * STAGE_LD + koff;
#pragma unroll
  for (int tn = 0; tn < 16; ++tn) {
    const _Float16* bp0 = Wt + (size_t)(tn * 16 + (lane & 15)) * 256 + koff;
    float8_t acc = {};
#pragma unroll
    for (int kk = 0; kk < 8; ++kk)
      acc = wmma_f16(load_frag_h(ap0 + kk * 32), load_frag_h(bp0 + kk * 32), acc);
    accs[tn] = acc;
  }
}

__global__ __launch_bounds__(128) void mlp_ln_k(
    const float* __restrict__ attn,
    const _Float16* __restrict__ W1T, const float* __restrict__ b1,
    const _Float16* __restrict__ W2T, const float* __restrict__ b2,
    const _Float16* __restrict__ W3T, const float* __restrict__ b3,
    const float* __restrict__ ln_g, const float* __restrict__ ln_b,
    float* __restrict__ outp) {
  __shared__ _Float16 stage[4 * 16 * STAGE_LD];

  const int lane = threadIdx.x & 31;
  const int wave = threadIdx.x >> 5;
  const int r0 = (blockIdx.x * 4 + wave) * 16;   // first of 16 rows
  _Float16* st = stage + wave * 16 * STAGE_LD;

  // stage attn rows (f32 -> f16) into LDS
  for (int it = 0; it < 32; ++it) {
    int id4 = lane + 32 * it;            // 1024 float4 chunks
    int m = id4 >> 6;
    int f = (id4 & 63) * 4;
    float4 x = *(const float4*)(attn + (size_t)(r0 + m) * 256 + f);
    _Float16* d = st + m * STAGE_LD + f;
    d[0] = (_Float16)x.x; d[1] = (_Float16)x.y;
    d[2] = (_Float16)x.z; d[3] = (_Float16)x.w;
  }
  __syncthreads();

  float8_t accs[16];
  const _Float16* Wts[2] = {W1T, W2T};
  const float*    bss[2] = {b1, b2};

  // two hidden layers with LeakyReLU(0.01), written back through LDS
  for (int l = 0; l < 2; ++l) {
    mlp_layer(st, Wts[l], lane, accs);
    const float* bias = bss[l];
#pragma unroll
    for (int tn = 0; tn < 16; ++tn) {
      int n = tn * 16 + (lane & 15);
      float bv = bias[n];
#pragma unroll
      for (int r = 0; r < 8; ++r) {
        int m = r + ((lane < 16) ? 0 : 8);
        float v = accs[tn][r] + bv;
        v = (v >= 0.f) ? v : ALPHA_MLP * v;
        st[m * STAGE_LD + n] = (_Float16)v;
      }
    }
    __syncthreads();
  }

  // final linear + bias + residual (f32)
  mlp_layer(st, W3T, lane, accs);
#pragma unroll
  for (int tn = 0; tn < 16; ++tn) {
    int n = tn * 16 + (lane & 15);
    float bv = b3[n];
#pragma unroll
    for (int r = 0; r < 8; ++r) {
      int m = r + ((lane < 16) ? 0 : 8);
      accs[tn][r] += bv + attn[(size_t)(r0 + m) * 256 + n];
    }
  }

  // LayerNorm over the feature dim: row m<8 lives in lanes 0..15, m>=8 in
  // lanes 16..31, so reduce across the 16-lane half with shfl_xor width 16.
  float sum[8], sq[8];
#pragma unroll
  for (int r = 0; r < 8; ++r) { sum[r] = 0.f; sq[r] = 0.f; }
#pragma unroll
  for (int tn = 0; tn < 16; ++tn)
#pragma unroll
    for (int r = 0; r < 8; ++r) {
      float v = accs[tn][r];
      sum[r] += v;
      sq[r]  += v * v;
    }
#pragma unroll
  for (int r = 0; r < 8; ++r) {
#pragma unroll
    for (int off = 1; off < 16; off <<= 1) {
      sum[r] += __shfl_xor(sum[r], off, 16);
      sq[r]  += __shfl_xor(sq[r],  off, 16);
    }
  }
  float mu[8], rstd[8];
#pragma unroll
  for (int r = 0; r < 8; ++r) {
    mu[r] = sum[r] * (1.f / 256.f);
    float var = sq[r] * (1.f / 256.f) - mu[r] * mu[r];
    rstd[r] = rsqrtf(var + LN_EPS);
  }
#pragma unroll
  for (int tn = 0; tn < 16; ++tn) {
    int n = tn * 16 + (lane & 15);
    float g = ln_g[n], bb = ln_b[n];
#pragma unroll
    for (int r = 0; r < 8; ++r) {
      int m = r + ((lane < 16) ? 0 : 8);
      float z = (accs[tn][r] - mu[r]) * rstd[r] * g + bb;
      outp[(size_t)(r0 + m) * 256 + n] = z;
    }
  }
}

// ---------------------------------------------------------------------------
// launch
// ---------------------------------------------------------------------------
extern "C" void kernel_launch(void* const* d_in, const int* in_sizes, int n_in,
                              void* d_out, int out_size, void* d_ws,
                              size_t ws_size, hipStream_t stream) {
  const float* h    = (const float*)d_in[0];
  const float* adj  = (const float*)d_in[1];
  const float* W    = (const float*)d_in[2];
  const float* av   = (const float*)d_in[3];
  const float* ln_g = (const float*)d_in[4];
  const float* ln_b = (const float*)d_in[5];
  const float* W1   = (const float*)d_in[6];
  const float* b1   = (const float*)d_in[7];
  const float* W2   = (const float*)d_in[8];
  const float* b2   = (const float*)d_in[9];
  const float* W3   = (const float*)d_in[10];
  const float* b3   = (const float*)d_in[11];
  float* outp = (float*)d_out;

  char* ws = (char*)d_ws;
  _Float16* WT   = (_Float16*)(ws + 0);
  _Float16* W1T  = (_Float16*)(ws + 131072);
  _Float16* W2T  = (_Float16*)(ws + 262144);
  _Float16* W3T  = (_Float16*)(ws + 393216);
  float*    Wh   = (float*)(ws + 524288);                // 16 MB
  float*    attn = (float*)(ws + 524288 + 16777216);     // 16 MB

  prep_weights_k<<<1024, 256, 0, stream>>>(W, W1, W2, W3, WT, W1T, W2T, W3T);
  gemm_wh_k<<<2048, 256, 0, stream>>>(h, WT, Wh);
  attn_k<<<256, 256, 0, stream>>>(Wh, adj, av, attn);
  mlp_ln_k<<<256, 128, 0, stream>>>(attn, W1T, b1, W2T, b2, W3T, b3,
                                    ln_g, ln_b, outp);
}
